// PatchedCausalSelfAttention_25761213841524
// MI455X (gfx1250) — matrix operand order repaired and checked by reading, not compiled, not run on hardware
//
#include <hip/hip_runtime.h>

typedef __attribute__((ext_vector_type(16))) _Float16 v16h;
typedef __attribute__((ext_vector_type(8)))  _Float16 v8h;
typedef __attribute__((ext_vector_type(8)))  float    v8f;
typedef __attribute__((ext_vector_type(4)))  unsigned int v4u;
typedef __attribute__((ext_vector_type(8)))  unsigned int v8u;

#define WMMA_F16(a,b,c) __builtin_amdgcn_wmma_f32_16x16x32_f16(false,(a),false,(b),(short)0,(c),false,false)

// fmax with DPP-shuffled partner (ctrl must be an immediate)
#define DPP_MAXF(v, ctrl) \
  fmaxf((v), __int_as_float(__builtin_amdgcn_update_dpp( \
      0, __float_as_int(v), (ctrl), 0xF, 0xF, true)))

// Butterfly max across the 16-lane DPP row (each score row lives in one row).
static __device__ inline float rowmax16(float v) {
  v = DPP_MAXF(v, 0xB1);   // quad_perm [1,0,3,2]  : xor 1
  v = DPP_MAXF(v, 0x4E);   // quad_perm [2,3,0,1]  : xor 2
  v = DPP_MAXF(v, 0x141);  // row_half_mirror      : xor 4 (quads uniform)
  v = DPP_MAXF(v, 0x140);  // row_mirror           : xor 8 (octets uniform)
  return v;
}

// A-matrix 16x32 f16 fragment (ISA 7.12.2): lane holds row M=lane%16;
// halves j=0..7 -> K = 8*laneHi + j ; j=8..15 -> K = 16 + 8*laneHi + (j-8)
static __device__ inline v16h make_afrag(const _Float16* p, int laneHi) {
  v8h lo = *(const v8h*)(p + 8*laneHi);
  v8h hi = *(const v8h*)(p + 16 + 8*laneHi);
  v16h a;
#pragma unroll
  for (int i = 0; i < 8; i++) { a[i] = lo[i]; a[i+8] = hi[i]; }
  return a;
}
// B-matrix 32x16 f16 fragment: lane holds col N=lane%16;
// halves j=0..7 -> K = 8*laneHi + j ; j=8..15 -> K = 16 + 8*laneHi + (j-8)
// p points at B^T row n (K-contiguous storage).
static __device__ inline v16h make_bfrag(const _Float16* p, int laneHi) {
  v8h lo = *(const v8h*)(p + 8*laneHi);
  v8h hi = *(const v8h*)(p + 16 + 8*laneHi);
  v16h b;
#pragma unroll
  for (int i = 0; i < 8; i++) { b[i] = lo[i]; b[i+8] = hi[i]; }
  return b;
}

__global__ void f32_to_f16_kernel(const float* __restrict__ src,
                                  _Float16* __restrict__ dst, int n) {
  int i = blockIdx.x * blockDim.x + threadIdx.x;
  int stride = gridDim.x * blockDim.x;
  for (; i < n; i += stride) dst[i] = (_Float16)src[i];
}

// Tiled WMMA GEMM: C[M,N] = A[M,K](f16) * Bw[K,N](f16) + bias, K-step 32.
// Block: 256 threads (8 waves). Block tile 128x64; each wave does 16x64.
// STAGE 0: A tile staged via per-lane async global->LDS (ASYNCcnt).
// STAGE 1: A tile staged via one Tensor Data Mover descriptor (TENSORcnt).
// B tile staged transposed by hand (DMA cannot reshape 2-byte elements).
// MODE 0: scatter QKV into Qh/Kh [B,H,S,hd] f16 and Vt [B,H,hd,S] f16.
// MODE 1: write f32 to outF[M,N].
template <int MODE, int STAGE>
__global__ __launch_bounds__(256)
void gemm16(const _Float16* __restrict__ A, const _Float16* __restrict__ Bw,
            const float* __restrict__ bias, int M, int N, int K,
            _Float16* __restrict__ Qh, _Float16* __restrict__ Kh,
            _Float16* __restrict__ Vt, float* __restrict__ outF) {
  __shared__ _Float16 As[128 * 32];
  __shared__ _Float16 Bs[64 * 32];   // transposed: Bs[n][k]
  const int tid = threadIdx.x;
  const int wave = tid >> 5, lane = tid & 31;
  const int laneHi = (lane >> 4) & 1, lane16 = lane & 15;
  const int m0 = blockIdx.y * 128, n0 = blockIdx.x * 64;

  v8f acc[4] = {};
  for (int kb = 0; kb < K; kb += 32) {
    if (STAGE == 0) { // per-lane async DMA: 128 rows x 32 halves into LDS
      int row = tid >> 1, off = (tid & 1) * 16;
      const _Float16* g = A + (size_t)(m0 + row) * K + kb + off;
      unsigned ldsa = (unsigned)(unsigned long long)&As[row * 32 + off];
      // IOFFSET is added to BOTH lds and global addresses (ISA ch.10 async)
      asm volatile(
          "global_load_async_to_lds_b128 %0, %1, off\n\t"
          "global_load_async_to_lds_b128 %0, %1, off offset:16"
          :: "v"(ldsa), "v"(g) : "memory");
      if (kb + 32 < K) __builtin_prefetch(g + 32, 0, 1);
    } else { // Tensor Data Mover: one 2-D tile descriptor, issued by wave 0
      if (__builtin_amdgcn_readfirstlane(tid >> 5) == 0) {
        unsigned long long ga =
            (unsigned long long)(const void*)(A + (size_t)m0 * K + kb);
        v4u g0;
        g0[0] = 1u;                                   // count=1, user mode
        g0[1] = (unsigned)(unsigned long long)&As[0]; // lds_addr (bytes)
        g0[2] = (unsigned)ga;                         // global_addr[31:0]
        g0[3] = (unsigned)(ga >> 32) | (2u << 30);    // addr[56:32] | type=2
        v8u g1;
        g1[0] = 0x10000u;      // workgroup_mask=0, data_size=1 (2 bytes)
        g1[1] = 32u << 16;     // tensor_dim0 = 32 elems (lo16)
        g1[2] = 128u << 16;    // tensor_dim0 hi=0 | tensor_dim1 = 128 (lo16)
        g1[3] = 32u << 16;     // tensor_dim1 hi=0 | tile_dim0 = 32
        g1[4] = 128u;          // tile_dim1 = 128 | tile_dim2 = 0
        g1[5] = (unsigned)K;   // tensor_dim0_stride lo32 (elems)
        g1[6] = 0u;            // stride hi | dim1_stride lo (unused, 2-D)
        g1[7] = 0u;
        asm volatile("tensor_load_to_lds %0, %1" :: "s"(g0), "s"(g1)
                     : "memory");
      }
      if (kb + 32 < K) {
        int row = tid >> 1, off = (tid & 1) * 16;
        __builtin_prefetch(A + (size_t)(m0 + row) * K + kb + 32 + off, 0, 1);
      }
    }
    { // stage B transposed: 32 k-rows x 64 cols -> Bs[n][k]
      int kr = tid >> 3, noff = (tid & 7) * 8;
      const _Float16* bg = Bw + (size_t)(kb + kr) * N + n0 + noff;
      v8h b = *(const v8h*)bg;
#pragma unroll
      for (int i = 0; i < 8; i++) Bs[(noff + i) * 32 + kr] = b[i];
      if (kb + 32 < K) __builtin_prefetch(bg + (size_t)32 * N, 0, 1);
    }
    if (STAGE == 0) asm volatile("s_wait_asynccnt 0" ::: "memory");
    else            __builtin_amdgcn_s_wait_tensorcnt(0);
    __syncthreads();
    v16h af = make_afrag(&As[(wave * 16 + lane16) * 32], laneHi);
#pragma unroll
    for (int t = 0; t < 4; t++) {
      v16h bf = make_bfrag(&Bs[(t * 16 + lane16) * 32], laneHi);
      acc[t] = WMMA_F16(af, bf, acc[t]);
    }
    __syncthreads();
  }
  // epilogue: C/D layout: lane, slot r -> row = r + 8*laneHi, col = lane%16
#pragma unroll
  for (int t = 0; t < 4; t++) {
    int n = n0 + t * 16 + lane16;
    float bv = bias[n];
#pragma unroll
    for (int r = 0; r < 8; r++) {
      int m = m0 + wave * 16 + r + 8 * laneHi;
      float val = acc[t][r] + bv;
      if (MODE == 0) {
        int which = n >> 10;            // 0:q 1:k 2:v
        int c2 = n & 1023;
        int h = c2 >> 6, d = c2 & 63;
        int b = m >> 11, s = m & 2047;
        size_t hb = (size_t)(b * 16 + h);
        _Float16 hv = (_Float16)val;
        if (which == 0)      Qh[(hb * 2048 + s) * 64 + d] = hv;
        else if (which == 1) Kh[(hb * 2048 + s) * 64 + d] = hv;
        else                 Vt[(hb * 64 + d) * 2048 + s] = hv;
      } else {
        outF[(size_t)m * N + n] = val;
      }
    }
  }
}

// Flash attention: one wave per (head, 16-query-row block).
// Row sums computed as P x ones via WMMA (replicated across lanes -> no
// cross-lane broadcast needed); row max via DPP butterfly.
__global__ __launch_bounds__(32)
void flash16(const _Float16* __restrict__ Qh, const _Float16* __restrict__ Kh,
             const _Float16* __restrict__ Vt, _Float16* __restrict__ X2) {
  __shared__ _Float16 P[16 * 32];
  const int lane = threadIdx.x;
  const int laneHi = (lane >> 4) & 1, lane16 = lane & 15;
  const int qblk = blockIdx.x;  // 0..127
  const int bh = blockIdx.y;    // 0..31
  const int S = 2048, HD = 64;
  const _Float16* Qb = Qh + (size_t)bh * S * HD;
  const _Float16* Kb = Kh + (size_t)bh * S * HD;
  const _Float16* Vb = Vt + (size_t)bh * HD * S;

  v16h qf[2];
#pragma unroll
  for (int c = 0; c < 2; c++)
    qf[c] = make_afrag(Qb + (size_t)(qblk * 16 + lane16) * HD + 32 * c, laneHi);

  v16h ones;
#pragma unroll
  for (int i = 0; i < 16; i++) ones[i] = (_Float16)1.0f;

  v8f o[4] = {};
  v8f lacc = {};          // l accumulator: rowsum(P) replicated over lanes
  float mrow[8];
#pragma unroll
  for (int r = 0; r < 8; r++) mrow[r] = -1e30f;

  const int qmaxrow = qblk * 16 + 15;
  for (int j = 0; j * 32 <= qmaxrow; ++j) {
    { // prefetch next key block of K and V into GL2
      int nk = (j + 1) * 32;
      if (nk <= qmaxrow) {
        __builtin_prefetch(Kb + (size_t)(nk + lane) * HD, 0, 1);
        __builtin_prefetch(Vb + (size_t)(2 * lane) * S + nk, 0, 1);
        __builtin_prefetch(Vb + (size_t)(2 * lane + 1) * S + nk, 0, 1);
      }
    }
    // scores for 32 keys (two 16-key tiles), hd=64 = two k-chunks
    v8f sc[2];
#pragma unroll
    for (int t = 0; t < 2; t++) {
      int key = j * 32 + t * 16 + lane16;
      v8f s = {};
#pragma unroll
      for (int c = 0; c < 2; c++) {
        v16h kf = make_bfrag(Kb + (size_t)key * HD + 32 * c, laneHi);
        s = WMMA_F16(qf[c], kf, s);
      }
      sc[t] = s;
    }
    // scale (1/sqrt(64)) + causal mask
#pragma unroll
    for (int t = 0; t < 2; t++) {
      int key = j * 32 + t * 16 + lane16;
#pragma unroll
      for (int r = 0; r < 8; r++) {
        int qrow = qblk * 16 + r + 8 * laneHi;
        float v = sc[t][r] * 0.125f;
        sc[t][r] = (key > qrow) ? -1e30f : v;
      }
    }
    // online softmax: row max (DPP), rescale, exponentiate
    float alpha[8];
#pragma unroll
    for (int r = 0; r < 8; r++) {
      float rm = rowmax16(fmaxf(sc[0][r], sc[1][r]));
      float mn = fmaxf(mrow[r], rm);
      alpha[r] = __expf(mrow[r] - mn);
      mrow[r] = mn;
      sc[0][r] = __expf(sc[0][r] - mn);
      sc[1][r] = __expf(sc[1][r] - mn);
    }
#pragma unroll
    for (int r = 0; r < 8; r++) {
      lacc[r] *= alpha[r];
#pragma unroll
      for (int t = 0; t < 4; t++) o[t][r] *= alpha[r];
    }
    // P (16x32) to LDS in row-major f16, then reload as A-fragment
#pragma unroll
    for (int t = 0; t < 2; t++)
#pragma unroll
      for (int r = 0; r < 8; r++)
        P[(r + 8 * laneHi) * 32 + t * 16 + lane16] = (_Float16)sc[t][r];
    __syncthreads();
    v16h pf = make_afrag(&P[lane16 * 32], laneHi);
    lacc = WMMA_F16(pf, ones, lacc);   // row sums, replicated across lanes
#pragma unroll
    for (int t = 0; t < 4; t++) {
      v16h vf = make_bfrag(Vb + (size_t)(t * 16 + lane16) * S + j * 32, laneHi);
      o[t] = WMMA_F16(pf, vf, o[t]);
    }
    __syncthreads();
  }
  // write O/l to X2 [b*2048+qrow][h*64+col] (f16)
  const int b = bh >> 4, h = bh & 15;
  float rl[8];
#pragma unroll
  for (int r = 0; r < 8; r++) rl[r] = 1.f / lacc[r];
#pragma unroll
  for (int t = 0; t < 4; t++) {
    int col = h * 64 + t * 16 + lane16;
#pragma unroll
    for (int r = 0; r < 8; r++) {
      int qrow = qblk * 16 + r + 8 * laneHi;
      X2[((size_t)(b * 2048 + qrow)) * 1024 + col] = (_Float16)(o[t][r] * rl[r]);
    }
  }
}

extern "C" void kernel_launch(void* const* d_in, const int* in_sizes, int n_in,
                              void* d_out, int out_size, void* d_ws, size_t ws_size,
                              hipStream_t stream) {
  (void)in_sizes; (void)n_in; (void)out_size; (void)ws_size;
  const float* hidden = (const float*)d_in[0];
  const float* attn_w = (const float*)d_in[1];
  const float* attn_b = (const float*)d_in[2];
  const float* proj_w = (const float*)d_in[3];
  const float* proj_b = (const float*)d_in[4];
  float* out = (float*)d_out;

  const int B = 2, S = 2048, D = 1024, H = 16, HD = 64;
  const int M = B * S;  // 4096

  _Float16* p = (_Float16*)d_ws;
  _Float16* Xf = p; p += (size_t)M * D;          // hidden f16
  _Float16* Wa = p; p += (size_t)D * 3 * D;      // c_attn_w f16
  _Float16* Wp = p; p += (size_t)D * D;          // c_proj_w f16
  _Float16* Qh = p; p += (size_t)B * H * S * HD; // [B,H,S,hd]
  _Float16* Kh = p; p += (size_t)B * H * S * HD; // [B,H,S,hd]
  _Float16* Vt = p; p += (size_t)B * H * HD * S; // [B,H,hd,S]
  _Float16* X2 = p; p += (size_t)M * D;          // attention output f16

  f32_to_f16_kernel<<<dim3(1024), 256, 0, stream>>>(hidden, Xf, M * D);
  f32_to_f16_kernel<<<dim3(1024), 256, 0, stream>>>(attn_w, Wa, D * 3 * D);
  f32_to_f16_kernel<<<dim3(512),  256, 0, stream>>>(proj_w, Wp, D * D);

  gemm16<0, 0><<<dim3(3 * D / 64, M / 128), 256, 0, stream>>>(
      Xf, Wa, attn_b, M, 3 * D, D, Qh, Kh, Vt, nullptr);

  flash16<<<dim3(S / 16, B * H), 32, 0, stream>>>(Qh, Kh, Vt, X2);

  gemm16<1, 1><<<dim3(D / 64, M / 128), 256, 0, stream>>>(
      X2, Wp, proj_b, M, D, D, nullptr, nullptr, nullptr, out);
}
